// ImprovedGraphSAGE_71433896067546
// MI455X (gfx1250) — matrix operand order repaired and checked by reading, not compiled, not run on hardware
//
#include <hip/hip_runtime.h>
#include <math.h>

// Problem constants (from the reference)
#define Nn 50000
#define Ee 800000
#define Hh 128
#define Cc 16

typedef __attribute__((ext_vector_type(16))) __bf16 v16bf;
typedef __attribute__((ext_vector_type(8)))  float  v8f;

union FragA { uint4 u[2]; v16bf v; };

__device__ __forceinline__ unsigned short f2bf_bits(float f) {
  unsigned u = __float_as_uint(f);
  u += 0x7fffu + ((u >> 16) & 1u);          // round-to-nearest-even
  return (unsigned short)(u >> 16);
}

__device__ __forceinline__ float gelu_erf(float v) {
  return 0.5f * v * (1.0f + erff(v * 0.70710678118654752f));
}

// Async copy of one 16B chunk per thread straight into LDS (gfx1250 async path,
// tracked by ASYNCcnt). Low 32 bits of a generic shared pointer == LDS byte
// offset on AMDGPU, which is what the VDST operand of the async op wants.
__device__ __forceinline__ void async_tile_copy16(void* lds_dst, const void* gsrc) {
  unsigned ldsa = (unsigned)(size_t)lds_dst;
  asm volatile("global_load_async_to_lds_b128 %0, %1, off\n\t"
               "s_wait_asynccnt 0"
               :
               : "v"(ldsa), "v"(gsrc)
               : "memory");
}

// ---------------------------------------------------------------------------
// Repack a row-major f32 weight (128 x ncols) into the per-lane bf16 B-fragment
// layout of v_wmma_f32_16x16x32_bf16:
//   out[((ntile*4 + kk)*32 + lane)*16 + j] = W[(kk*32 + (j>=8?16:0) + (lane>=16?8:0) + (j&7))*ncols
//                                              + ntile*16 + (lane&15)]
// ---------------------------------------------------------------------------
__global__ void pack_w(const float* __restrict__ W, unsigned short* __restrict__ out,
                       int ncols) {
  int tid = blockIdx.x * 256 + threadIdx.x;
  int total = (ncols >> 4) * 4 * 32 * 16;
  if (tid >= total) return;
  int j     = tid & 15;
  int lane  = (tid >> 4) & 31;
  int kk    = (tid >> 9) & 3;
  int ntile = tid >> 11;
  int k   = kk * 32 + ((j >> 3) << 4) + ((lane >> 4) << 3) + (j & 7);
  int col = ntile * 16 + (lane & 15);
  out[tid] = f2bf_bits(W[k * ncols + col]);
}

// h0 = bf16(x * importance); also deg[i] = 1.0 (self-loop)
__global__ void feat_init(const float* __restrict__ x, const float* __restrict__ imp,
                          unsigned short* __restrict__ hbf, float* __restrict__ deg) {
  int tid = blockIdx.x * 256 + threadIdx.x;
  if (tid >= Nn * Hh) return;
  int f = tid & 127;
  hbf[tid] = f2bf_bits(x[tid] * imp[f]);
  if (f == 0) deg[tid >> 7] = 1.0f;
}

__global__ void deg_scatter(const long long* __restrict__ ei, float* __restrict__ deg) {
  int e = blockIdx.x * 256 + threadIdx.x;
  if (e >= Ee) return;
  atomicAdd(deg + (int)ei[Ee + e], 1.0f);
}

__global__ void make_dinv(float* __restrict__ d) {
  int i = blockIdx.x * 256 + threadIdx.x;
  if (i >= Nn) return;
  d[i] = 1.0f / sqrtf(d[i]);
}

// ---------------------------------------------------------------------------
// Conv GEMM: y = A @ W (bf16 WMMA, fp32 acc). Epilogue writes
//   ybf = bf16(y)                         (edge-gather source)
//   acc = dinv[i]^2 * y + bias            (self-loop term + bias pre-folded)
// Block = 256 thr = 8 waves; block owns one 16-row tile, wave w owns cols [16w,16w+16)
// ---------------------------------------------------------------------------
__global__ void gemm_conv(const unsigned short* __restrict__ A,
                          const unsigned short* __restrict__ Wpk,
                          const float* __restrict__ bias,
                          const float* __restrict__ dinv,
                          unsigned short* __restrict__ ybf,
                          float* __restrict__ acc) {
  __shared__ __align__(16) unsigned short As[16][128];
  const int tile = blockIdx.x;
  const int t = threadIdx.x;
  const int wave = t >> 5, lane = t & 31;
  // async 4KB tile stage: 256 threads x 16B (N = 3125*16 exactly -> no guards)
  async_tile_copy16(reinterpret_cast<uint4*>(&As[0][0]) + t,
                    reinterpret_cast<const uint4*>(A + tile * 16 * Hh) + t);
  __syncthreads();

  const int m = lane & 15;
  const int half = lane >> 4;
  v8f c = {};
#pragma unroll
  for (int kk = 0; kk < 4; ++kk) {
    FragA fa;
    const uint4* rp = reinterpret_cast<const uint4*>(&As[m][kk * 32 + (half << 3)]);
    fa.u[0] = rp[0];        // K chunk [base+8h, base+8h+8)
    fa.u[1] = rp[2];        // K chunk [base+8h+16, base+8h+24)
    v16bf b = *reinterpret_cast<const v16bf*>(Wpk + (((wave * 4 + kk) * 32 + lane) << 4));
    c = __builtin_amdgcn_wmma_f32_16x16x32_bf16(false, fa.v, false, b,
                                                (short)0, c, false, false);
  }
  const int col = wave * 16 + m;
#pragma unroll
  for (int r = 0; r < 8; ++r) {
    int row = tile * 16 + (half << 3) + r;
    float y = c[r];
    float dv = dinv[row];
    int off = row * Hh + col;
    ybf[off] = f2bf_bits(y);
    acc[off] = dv * dv * y + bias[col];
  }
}

// Dense layer: out = bf16(gelu(A @ W + bias))
__global__ void gemm_dense(const unsigned short* __restrict__ A,
                           const unsigned short* __restrict__ Wpk,
                           const float* __restrict__ bias,
                           unsigned short* __restrict__ outbf) {
  __shared__ __align__(16) unsigned short As[16][128];
  const int tile = blockIdx.x;
  const int t = threadIdx.x;
  const int wave = t >> 5, lane = t & 31;
  async_tile_copy16(reinterpret_cast<uint4*>(&As[0][0]) + t,
                    reinterpret_cast<const uint4*>(A + tile * 16 * Hh) + t);
  __syncthreads();

  const int m = lane & 15;
  const int half = lane >> 4;
  v8f c = {};
#pragma unroll
  for (int kk = 0; kk < 4; ++kk) {
    FragA fa;
    const uint4* rp = reinterpret_cast<const uint4*>(&As[m][kk * 32 + (half << 3)]);
    fa.u[0] = rp[0];
    fa.u[1] = rp[2];
    v16bf b = *reinterpret_cast<const v16bf*>(Wpk + (((wave * 4 + kk) * 32 + lane) << 4));
    c = __builtin_amdgcn_wmma_f32_16x16x32_bf16(false, fa.v, false, b,
                                                (short)0, c, false, false);
  }
  const int col = wave * 16 + m;
#pragma unroll
  for (int r = 0; r < 8; ++r) {
    int row = tile * 16 + (half << 3) + r;
    outbf[row * Hh + col] = f2bf_bits(gelu_erf(c[r] + bias[col]));
  }
}

// Head: out = A @ lw2 + lb2   (C = 16 -> one column tile; one wave = one row tile)
__global__ void gemm_out(const unsigned short* __restrict__ A,
                         const unsigned short* __restrict__ Wpk,
                         const float* __restrict__ bias,
                         float* __restrict__ out) {
  const int t = threadIdx.x;
  const int wave = t >> 5, lane = t & 31;
  const int tile = blockIdx.x * 8 + wave;
  if (tile >= Nn / 16) return;
  const int m = lane & 15;
  const int half = lane >> 4;
  v8f c = {};
#pragma unroll
  for (int kk = 0; kk < 4; ++kk) {
    FragA fa;
    const uint4* rp = reinterpret_cast<const uint4*>(
        A + (tile * 16 + m) * Hh + kk * 32 + (half << 3));
    fa.u[0] = rp[0];
    fa.u[1] = rp[2];
    v16bf b = *reinterpret_cast<const v16bf*>(Wpk + ((kk * 32 + lane) << 4));
    c = __builtin_amdgcn_wmma_f32_16x16x32_bf16(false, fa.v, false, b,
                                                (short)0, c, false, false);
  }
#pragma unroll
  for (int r = 0; r < 8; ++r) {
    int row = tile * 16 + (half << 3) + r;
    out[row * Cc + m] = c[r] + bias[m];
  }
}

// Edge messages: acc[col] += dinv[row]*dinv[col] * y[row]
// one wave per edge, 4 features per lane; bf16 gather, f32 hardware atomics
__global__ void edge_msg(const long long* __restrict__ ei,
                         const float* __restrict__ dinv,
                         const unsigned short* __restrict__ ybf,
                         float* __restrict__ acc) {
  int tid = blockIdx.x * 256 + threadIdx.x;
  int e = tid >> 5;
  if (e >= Ee) return;
  int lane = tid & 31;
  int r = (int)ei[e];
  int c = (int)ei[Ee + e];
  float nrm = dinv[r] * dinv[c];
  uint2 raw = *reinterpret_cast<const uint2*>(ybf + r * Hh + lane * 4);
  float* ap = acc + c * Hh + lane * 4;
  atomicAdd(ap + 0, nrm * __uint_as_float((raw.x & 0xffffu) << 16));
  atomicAdd(ap + 1, nrm * __uint_as_float(raw.x & 0xffff0000u));
  atomicAdd(ap + 2, nrm * __uint_as_float((raw.y & 0xffffu) << 16));
  atomicAdd(ap + 3, nrm * __uint_as_float(raw.y & 0xffff0000u));
}

// h_next = bf16(gelu(acc))
__global__ void act_gelu(const float* __restrict__ acc, unsigned short* __restrict__ hbf) {
  int tid = blockIdx.x * 256 + threadIdx.x;
  if (tid >= Nn * Hh) return;
  hbf[tid] = f2bf_bits(gelu_erf(acc[tid]));
}

// ---------------------------------------------------------------------------
extern "C" void kernel_launch(void* const* d_in, const int* in_sizes, int n_in,
                              void* d_out, int out_size, void* d_ws, size_t ws_size,
                              hipStream_t stream) {
  const float*     x   = (const float*)d_in[0];
  const long long* ei  = (const long long*)d_in[1];   // int64 edge_index [2,E]
  const float*     imp = (const float*)d_in[2];
  const float*     W1  = (const float*)d_in[3];
  const float*     b1  = (const float*)d_in[4];
  const float*     W2  = (const float*)d_in[5];
  const float*     b2  = (const float*)d_in[6];
  const float*     W3  = (const float*)d_in[7];
  const float*     b3  = (const float*)d_in[8];
  const float*     lw1 = (const float*)d_in[9];
  const float*     lb1 = (const float*)d_in[10];
  const float*     lw2 = (const float*)d_in[11];
  const float*     lb2 = (const float*)d_in[12];
  float* out = (float*)d_out;

  char* ws = (char*)d_ws;
  size_t off = 0;
  auto alloc = [&](size_t bytes) -> void* {
    void* p = ws + off;
    off += (bytes + 255) & ~(size_t)255;
    return p;
  };
  float*          dinv = (float*)alloc((size_t)Nn * 4);
  float*          acc  = (float*)alloc((size_t)Nn * Hh * 4);
  unsigned short* hbf  = (unsigned short*)alloc((size_t)Nn * Hh * 2);
  unsigned short* hbf2 = (unsigned short*)alloc((size_t)Nn * Hh * 2);
  unsigned short* ybf  = (unsigned short*)alloc((size_t)Nn * Hh * 2);
  unsigned short* w1p  = (unsigned short*)alloc(16384 * 2);
  unsigned short* w2p  = (unsigned short*)alloc(16384 * 2);
  unsigned short* w3p  = (unsigned short*)alloc(16384 * 2);
  unsigned short* lw1p = (unsigned short*)alloc(16384 * 2);
  unsigned short* lw2p = (unsigned short*)alloc(2048 * 2);

  // weight repack (once per launch; tiny)
  pack_w<<<64, 256, 0, stream>>>(W1, w1p, 128);
  pack_w<<<64, 256, 0, stream>>>(W2, w2p, 128);
  pack_w<<<64, 256, 0, stream>>>(W3, w3p, 128);
  pack_w<<<64, 256, 0, stream>>>(lw1, lw1p, 128);
  pack_w<<<8, 256, 0, stream>>>(lw2, lw2p, 16);

  // normalization (graph is identical across layers -> compute once)
  feat_init<<<(Nn * Hh + 255) / 256, 256, 0, stream>>>(x, imp, hbf, dinv);
  deg_scatter<<<(Ee + 255) / 256, 256, 0, stream>>>(ei, dinv);
  make_dinv<<<(Nn + 255) / 256, 256, 0, stream>>>(dinv);

  const int ntiles = Nn / 16;                 // 3125 exactly
  const int eblocks = (Ee * 32 + 255) / 256;  // 100000
  const int ablocks = (Nn * Hh + 255) / 256;  // 25000

  // layer 1
  gemm_conv<<<ntiles, 256, 0, stream>>>(hbf, w1p, b1, dinv, ybf, acc);
  edge_msg<<<eblocks, 256, 0, stream>>>(ei, dinv, ybf, acc);
  act_gelu<<<ablocks, 256, 0, stream>>>(acc, hbf);
  // layer 2
  gemm_conv<<<ntiles, 256, 0, stream>>>(hbf, w2p, b2, dinv, ybf, acc);
  edge_msg<<<eblocks, 256, 0, stream>>>(ei, dinv, ybf, acc);
  act_gelu<<<ablocks, 256, 0, stream>>>(acc, hbf);
  // layer 3
  gemm_conv<<<ntiles, 256, 0, stream>>>(hbf, w3p, b3, dinv, ybf, acc);
  edge_msg<<<eblocks, 256, 0, stream>>>(ei, dinv, ybf, acc);
  act_gelu<<<ablocks, 256, 0, stream>>>(acc, hbf);

  // dense + head
  gemm_dense<<<ntiles, 256, 0, stream>>>(hbf, lw1p, lb1, hbf2);
  gemm_out<<<(ntiles + 7) / 8, 256, 0, stream>>>(hbf2, lw2p, lb2, out);
}
